// LengthRegulator_32418413150800
// MI455X (gfx1250) — compile-verified, matrix operand order checked by reading, not run on hardware
//
#include <hip/hip_runtime.h>
#include <hip/hip_bf16.h>

#define BB 32
#define TT 512
#define DD 512
#define FF 256
#define MEL 4096
#define EPSV 1e-5f

typedef __attribute__((ext_vector_type(16))) __bf16 v16bf;
typedef __attribute__((ext_vector_type(8)))  float  v8f;
typedef __attribute__((ext_vector_type(4)))  float  f4;

__device__ __forceinline__ unsigned short f2bf(float f) {
    unsigned int u = __float_as_uint(f);
    u += 0x7FFFu + ((u >> 16) & 1u);   // round-to-nearest-even
    return (unsigned short)(u >> 16);
}

// ---------------- prep kernels ----------------

__global__ void cvt_x_bf16(const float* __restrict__ src,
                           unsigned short* __restrict__ dst, int n2) {
    int i = blockIdx.x * blockDim.x + threadIdx.x;
    if (i < n2) {
        float a = src[2 * i], b = src[2 * i + 1];
        reinterpret_cast<unsigned int*>(dst)[i] =
            (unsigned int)f2bf(a) | ((unsigned int)f2bf(b) << 16);
    }
}

// w[k][d][f] (f32) -> wt[k][f][d] (bf16)
__global__ void wtrans_bf16(const float* __restrict__ w,
                            unsigned short* __restrict__ wt,
                            int D, int F, int n) {
    int i = blockIdx.x * blockDim.x + threadIdx.x;
    if (i < n) {
        int k = i / (D * F);
        int rem = i - k * D * F;
        int f = rem / D;
        int d = rem - f * D;
        wt[i] = f2bf(w[(k * D + d) * F + f]);
    }
}

__global__ void cumsum_k(const int* __restrict__ target, int* __restrict__ cum) {
    int b = threadIdx.x;
    if (b < BB) {
        int s = 0;
        for (int t = 0; t < TT; ++t) { s += target[b * TT + t]; cum[b * TT + t] = s; }
    }
}

// idxm[b][p] = searchsorted_right(cum[b], p) clipped, or -1 if p >= total
__global__ void build_idx(const int* __restrict__ cum, int* __restrict__ idxm) {
    int gid = blockIdx.x * blockDim.x + threadIdx.x;
    if (gid < BB * MEL) {
        int b = gid >> 12;
        int p = gid & (MEL - 1);
        const int* c = cum + b * TT;
        int total = c[TT - 1];
        int lo = 0, hi = TT;
        while (lo < hi) {
            int mid = (lo + hi) >> 1;
            if (c[mid] <= p) lo = mid + 1; else hi = mid;
        }
        int idx = lo < TT ? lo : TT - 1;
        idxm[gid] = (p < total) ? idx : -1;
    }
}

// ---------------- conv + LN + ReLU (WMMA bf16) ----------------
// One block = 512 threads = 16 wave32s; covers 16 time rows x all 256 F.
// Wave w computes the 16x16 tile at f0 = 16*w via v_wmma_f32_16x16x32_bf16,
// K-loop = 3 (conv taps) x DIN/32.
template <int DIN, bool DOT>
__global__ void __launch_bounds__(512)
conv_ln_relu(const unsigned short* __restrict__ in,   // (B,T,DIN) bf16
             const unsigned short* __restrict__ wt,   // (3,FF,DIN) bf16
             const float* __restrict__ bias,
             const float* __restrict__ gamma,
             const float* __restrict__ beta,
             unsigned short* __restrict__ out,        // (B,T,FF) bf16 (!DOT)
             const float* __restrict__ wl,            // (FF) (DOT)
             const float* __restrict__ bl,            // (1)  (DOT)
             float* __restrict__ dur)                 // (B,T) (DOT)
{
    __shared__ __align__(32) unsigned short sA[18 * DIN]; // halo strip t0-1..t0+16
    __shared__ float sH[16 * FF];
    __shared__ float sMu[16], sRs[16];

    const int b   = blockIdx.y;
    const int t0  = blockIdx.x * 16;
    const int tid = threadIdx.x;

    // stage input strip (zero outside [0,T))
    const int row32 = DIN / 2;
    for (int i = tid; i < 18 * row32; i += 512) {
        int r = i / row32, c = i - r * row32;
        int t = t0 - 1 + r;
        unsigned int v = 0u;
        if (t >= 0 && t < TT)
            v = reinterpret_cast<const unsigned int*>(
                    in + (size_t)(b * TT + t) * DIN)[c];
        reinterpret_cast<unsigned int*>(sA)[i] = v;
    }
    __syncthreads();

    const int wave  = tid >> 5;
    const int lane  = tid & 31;
    const int mrow  = lane & 15;            // A: row M ; B: col N
    const int khalf = (lane < 16) ? 0 : 16; // contiguous K half: k = 16*(lane/16) + s (A and B alike)
    const int f0    = wave * 16;

    v8f c = {};
    for (int k = 0; k < 3; ++k) {
        const unsigned short* ap = sA + (mrow + k) * DIN + khalf;
        const unsigned short* bp = wt + (size_t)(k * FF + f0 + mrow) * DIN + khalf;
#pragma unroll
        for (int d0 = 0; d0 < DIN; d0 += 32) {
            v16bf av = *reinterpret_cast<const v16bf*>(ap + d0);
            v16bf bv = *reinterpret_cast<const v16bf*>(bp + d0);
            c = __builtin_amdgcn_wmma_f32_16x16x32_bf16(
                    false, av, false, bv, (short)0, c, false, false);
        }
    }

    // scatter C tile (+bias) into sH[16][FF]
    {
        const float bv   = bias[f0 + mrow];
        const int Mbase  = (lane < 16) ? 0 : 8;
#pragma unroll
        for (int r = 0; r < 8; ++r)
            sH[(Mbase + r) * FF + f0 + mrow] = c[r] + bv;
    }
    __syncthreads();

    // LayerNorm stats: wave w reduces row w across F=256
    {
        const int row = wave;
        float s = 0.f, s2 = 0.f;
#pragma unroll
        for (int j = 0; j < 8; ++j) {
            float v = sH[row * FF + lane + 32 * j];
            s += v; s2 += v * v;
        }
#pragma unroll
        for (int m = 16; m >= 1; m >>= 1) {
            s  += __shfl_xor(s,  m, 32);
            s2 += __shfl_xor(s2, m, 32);
        }
        if (lane == 0) {
            float mu  = s * (1.f / FF);
            float var = s2 * (1.f / FF) - mu * mu;
            sMu[row] = mu;
            sRs[row] = rsqrtf(var + EPSV);
        }
    }
    __syncthreads();

    if (!DOT) {
        // normalize + relu -> bf16 (packed pair stores)
        for (int i = tid; i < 16 * (FF / 2); i += 512) {
            int r  = i / (FF / 2);
            int c2 = (i - r * (FF / 2)) * 2;
            float mu = sMu[r], rs = sRs[r];
            float v0 = (sH[r * FF + c2]     - mu) * rs * gamma[c2]     + beta[c2];
            float v1 = (sH[r * FF + c2 + 1] - mu) * rs * gamma[c2 + 1] + beta[c2 + 1];
            v0 = fmaxf(v0, 0.f); v1 = fmaxf(v1, 0.f);
            reinterpret_cast<unsigned int*>(
                out + (size_t)(b * TT + t0 + r) * FF)[c2 >> 1] =
                (unsigned int)f2bf(v0) | ((unsigned int)f2bf(v1) << 16);
        }
    } else {
        // normalize + relu in-place, then dot with wl -> dur_pred
        for (int i = tid; i < 16 * FF; i += 512) {
            int r = i >> 8, cc = i & (FF - 1);
            float v = (sH[i] - sMu[r]) * sRs[r] * gamma[cc] + beta[cc];
            sH[i] = fmaxf(v, 0.f);
        }
        __syncthreads();
        const int row = wave;
        float s = 0.f;
#pragma unroll
        for (int j = 0; j < 8; ++j)
            s += sH[row * FF + lane + 32 * j] * wl[lane + 32 * j];
#pragma unroll
        for (int m = 16; m >= 1; m >>= 1)
            s += __shfl_xor(s, m, 32);
        if (lane == 0)
            dur[b * TT + t0 + row] = fmaxf(s + bl[0], 0.f);
    }
}

// ---------------- length-regulator gather (BW-bound) ----------------
// 256 threads / block, 2 output rows per block, 128-bit loads + NT stores.
__global__ void __launch_bounds__(256)
gather_k(const float* __restrict__ x, const int* __restrict__ idxm,
         float* __restrict__ out) {
    __shared__ int sIdx[2];
    const int b   = blockIdx.y;
    const int p0  = blockIdx.x * 2;
    const int tid = threadIdx.x;
    if (tid < 2) sIdx[tid] = idxm[b * MEL + p0 + tid];
    __syncthreads();
    const int sub = tid >> 7;
    const int l   = tid & 127;
    const int p   = p0 + sub;
    const int idx = sIdx[sub];
    f4 v = {};
    if (idx >= 0)
        v = reinterpret_cast<const f4*>(x + (size_t)(b * TT + idx) * DD)[l];
    __builtin_nontemporal_store(
        v, reinterpret_cast<f4*>(out + (size_t)(b * MEL + p) * DD) + l);
}

// ---------------- launch ----------------

extern "C" void kernel_launch(void* const* d_in, const int* in_sizes, int n_in,
                              void* d_out, int out_size, void* d_ws, size_t ws_size,
                              hipStream_t stream) {
    const float* x      = (const float*)d_in[0];
    const int*   target = (const int*)d_in[1];
    const float* w1     = (const float*)d_in[3];
    const float* b1     = (const float*)d_in[4];
    const float* g1     = (const float*)d_in[5];
    const float* be1    = (const float*)d_in[6];
    const float* w2     = (const float*)d_in[7];
    const float* b2     = (const float*)d_in[8];
    const float* g2     = (const float*)d_in[9];
    const float* be2    = (const float*)d_in[10];
    const float* wl     = (const float*)d_in[11];
    const float* bl     = (const float*)d_in[12];

    float* out = (float*)d_out;
    float* dur = out + (size_t)BB * MEL * DD;

    char* ws = (char*)d_ws;
    unsigned short* xb  = (unsigned short*)(ws);              // 16 MB
    unsigned short* h1  = (unsigned short*)(ws + 16777216);   // 8 MB
    unsigned short* w1t = (unsigned short*)(ws + 25165824);   // 768 KB
    unsigned short* w2t = (unsigned short*)(ws + 25952256);   // 384 KB
    int* cum            = (int*)(ws + 26345472);              // 64 KB
    int* idxm           = (int*)(ws + 26411008);              // 512 KB

    cvt_x_bf16<<<(BB * TT * DD / 2 + 255) / 256, 256, 0, stream>>>(x, xb, BB * TT * DD / 2);
    wtrans_bf16<<<(3 * DD * FF + 255) / 256, 256, 0, stream>>>(w1, w1t, DD, FF, 3 * DD * FF);
    wtrans_bf16<<<(3 * FF * FF + 255) / 256, 256, 0, stream>>>(w2, w2t, FF, FF, 3 * FF * FF);
    cumsum_k<<<1, 32, 0, stream>>>(target, cum);
    build_idx<<<(BB * MEL + 255) / 256, 256, 0, stream>>>(cum, idxm);

    conv_ln_relu<DD, false><<<dim3(TT / 16, BB), 512, 0, stream>>>(
        xb, w1t, b1, g1, be1, h1, nullptr, nullptr, nullptr);
    conv_ln_relu<FF, true><<<dim3(TT / 16, BB), 512, 0, stream>>>(
        h1, w2t, b2, g2, be2, nullptr, wl, bl, dur);

    gather_k<<<dim3(MEL / 2, BB), 256, 0, stream>>>(x, idxm, out);
}
